// ThermalDiffusionConv_77704548319355
// MI455X (gfx1250) — matrix operand order repaired and checked by reading, not compiled, not run on hardware
//
#include <hip/hip_runtime.h>
#include <hip/hip_bf16.h>
#include <math.h>

typedef __attribute__((ext_vector_type(16))) _Float16 v16h;
typedef __attribute__((ext_vector_type(8)))  _Float16 v8h;
typedef __attribute__((ext_vector_type(8)))  float    v8f;
typedef __attribute__((ext_vector_type(4)))  int      v4i;

#define AS1 __attribute__((address_space(1)))
#define AS3 __attribute__((address_space(3)))

#define BATCH 16
#define CIN   64
#define COUT  64
#define HH    128
#define WW    128
#define KTOT  576          // 9 taps * 64 channels, k = tap*64 + c
#define KSTEPS 18          // 576 / 32

#ifndef __has_builtin
#define __has_builtin(x) 0
#endif

#if __has_builtin(__builtin_amdgcn_global_load_async_to_lds_b128)
#define ASYNC_LDS 1
#else
#define ASYNC_LDS 0
#endif

__device__ __forceinline__ void wait_async0() {
#if __has_builtin(__builtin_amdgcn_s_wait_asynccnt)
    __builtin_amdgcn_s_wait_asynccnt(0);
#else
    asm volatile("s_wait_asynccnt 0x0" ::: "memory");
#endif
}

// ---------------------------------------------------------------------------
// prep: f32 (o, c, 3, 3) weights -> f16 [o][k] with k = tap*64 + c ; zero GAP
// ---------------------------------------------------------------------------
__global__ __launch_bounds__(256)
void prep_kernel(const float* __restrict__ dn_w1,
                 const float* __restrict__ bc_w1,
                 const float* __restrict__ bc_w2,
                 _Float16* __restrict__ w16,     // 3 sets of [64][576]
                 float* __restrict__ gap)        // [16][64]
{
    int idx = blockIdx.x * 256 + threadIdx.x;
    if (idx < BATCH * COUT) gap[idx] = 0.0f;
    if (idx < 3 * COUT * KTOT) {
        int set = idx / (COUT * KTOT);
        int rem = idx % (COUT * KTOT);
        int o = rem / KTOT;
        int k = rem % KTOT;
        int r = k >> 6;          // tap 0..8
        int c = k & 63;          // input channel
        const float* src = (set == 0) ? dn_w1 : (set == 1) ? bc_w1 : bc_w2;
        w16[idx] = (_Float16)src[o * KTOT + c * 9 + r];
    }
}

// ---------------------------------------------------------------------------
// convA: fused dn-conv1(ReLU->GAP) + bc-conv1(tanh->t1) + xs row sum
// one block = one (batch, row); 8 waves; WMMA implicit GEMM
// t1 is written CHANNEL-INNERMOST: (b, y, x, c) f16  -> b128 stores here,
// b128 async LDS loads in convC.
// ---------------------------------------------------------------------------
__global__ __launch_bounds__(256)
void convA_kernel(const float* __restrict__ x,
                  const float* __restrict__ dn_b1,
                  const float* __restrict__ bc_b1,
                  const _Float16* __restrict__ w16,  // set0=dn_w1, set1=bc_w1
                  _Float16* __restrict__ t1,         // (b,128,128,64) f16
                  float* __restrict__ xs,            // (b,128,128) f32
                  float* __restrict__ gap)           // (b,64) f32
{
    __shared__ _Float16 sm[3 * 130 * 64];   // [row][col(+halo)][channel]

    const int b   = blockIdx.y;
    const int y   = blockIdx.x;
    const int tid = threadIdx.x;
    const int wid = tid >> 5;
    const int lane = tid & 31;

    // ---- stage input halo tile, f32 -> f16, channel-innermost ----
    for (int p = wid; p < 192; p += 8) {          // p = row*64 + c
        int row = p >> 6, c = p & 63;
        int yy = y + row - 1;
        const float* src = x + (((size_t)(b * CIN + c) * HH + yy) * WW);
        for (int col = lane; col < 130; col += 32) {
            float v = 0.0f;
            int xx = col - 1;
            if (yy >= 0 && yy < HH && xx >= 0 && xx < WW) v = src[xx];
            sm[(row * 130 + col) * 64 + c] = (_Float16)v;
        }
    }
    __syncthreads();

    // ---- xs = sum over channels (center row) ----
    if (tid < WW) {
        float s = 0.0f;
        const _Float16* p = &sm[(130 + tid + 1) * 64];
        #pragma unroll 8
        for (int c = 0; c < 64; ++c) s += (float)p[c];
        xs[((size_t)b * HH + y) * WW + tid] = s;
    }

    // ---- WMMA GEMM: D(16 oc x 16 px) = A(16x32) * B(32x16), K = 576 ----
    const int mbase   = (wid & 3) * 16;     // out-channel tile
    const int nhalf   = (wid >> 2) * 64;    // pixel half (0 or 64)
    const int halfsel = lane >> 4;          // 0: lanes 0-15, 1: lanes 16-31
    const int nloc    = lane & 15;

    v8f acc[2][4] = {};                     // [dn|bc][pixel subtile]

    const int o  = mbase + nloc;
    const int K0 = halfsel ? 16 : 0;
    const _Float16* wdn = w16 + (size_t)o * KTOT;
    const _Float16* wbc = w16 + (size_t)COUT * KTOT + (size_t)o * KTOT;
    __builtin_prefetch(wdn, 0, 1);
    __builtin_prefetch(wbc, 0, 1);

    #pragma unroll 2
    for (int s = 0; s < KSTEPS; ++s) {
        const int kb = s * 32;
        const int r  = s >> 1;              // tap index (uniform)
        const int dy = r / 3, dx = r % 3;
        const int c0 = ((s & 1) << 5) + (halfsel << 4);  // channel base for B

        union { v16h v; v8h h[2]; } adn, abc;
        adn.h[0] = *(const v8h*)(wdn + kb + K0);
        adn.h[1] = *(const v8h*)(wdn + kb + K0 + 8);
        abc.h[0] = *(const v8h*)(wbc + kb + K0);
        abc.h[1] = *(const v8h*)(wbc + kb + K0 + 8);

        #pragma unroll
        for (int t = 0; t < 4; ++t) {
            int px = nhalf + t * 16 + nloc;
            const v16h bfrag = *(const v16h*)(&sm[(dy * 130 + px + dx) * 64 + c0]);
            acc[0][t] = __builtin_amdgcn_wmma_f32_16x16x32_f16(
                false, adn.v, false, bfrag, (short)0, acc[0][t], false, false);
            acc[1][t] = __builtin_amdgcn_wmma_f32_16x16x32_f16(
                false, abc.v, false, bfrag, (short)0, acc[1][t], false, false);
        }
    }

    // ---- epilogue ----
    const int chbase = mbase + halfsel * 8;   // 8 consecutive channels per lane
    #pragma unroll
    for (int j = 0; j < 8; ++j) {
        int ch = chbase + j;
        // dn: ReLU then GAP partial -> shuffle reduce -> atomic add
        float bd = dn_b1[ch];
        float s = 0.0f;
        #pragma unroll
        for (int t = 0; t < 4; ++t) s += fmaxf(acc[0][t][j] + bd, 0.0f);
        s += __shfl_xor(s, 1, 32);
        s += __shfl_xor(s, 2, 32);
        s += __shfl_xor(s, 4, 32);
        s += __shfl_xor(s, 8, 32);
        if (nloc == 0) atomicAdd(&gap[b * COUT + ch], s);
    }
    // bc: tanh -> t1 (channel-innermost, packed b128 stores)
    float bb[8];
    #pragma unroll
    for (int j = 0; j < 8; ++j) bb[j] = bc_b1[chbase + j];
    _Float16* trow = t1 + ((size_t)(b * HH + y) * WW) * 64;
    #pragma unroll
    for (int t = 0; t < 4; ++t) {
        int px = nhalf + t * 16 + nloc;
        v8h pk;
        #pragma unroll
        for (int j = 0; j < 8; ++j)
            pk[j] = (_Float16)tanhf(acc[1][t][j] + bb[j]);
        *(v8h*)(trow + (size_t)px * 64 + chbase) = pk;
    }
}

// ---------------------------------------------------------------------------
// head: GAP mean -> FC(64->512) -> sigmoid -> thermal 3x3 kernels (b,64,9)
// ---------------------------------------------------------------------------
__global__ __launch_bounds__(256)
void head_kernel(const float* __restrict__ dn_w2,   // (512,64)
                 const float* __restrict__ dn_b2,   // (512,)
                 const float* __restrict__ gap,     // (b,64) summed
                 float* __restrict__ ker)           // (b,64,9)
{
    __shared__ float sy[512];
    __shared__ float sg[64];
    const int b = blockIdx.x, tid = threadIdx.x;
    if (tid < 64) sg[tid] = gap[b * 64 + tid] * (1.0f / (HH * WW));
    __syncthreads();
    for (int o = tid; o < 512; o += 256) {
        float a = dn_b2[o];
        const float* wr = dn_w2 + o * 64;
        #pragma unroll 8
        for (int c = 0; c < 64; ++c) a += wr[c] * sg[c];
        sy[o] = 1.0f / (1.0f + __expf(-a));
    }
    __syncthreads();
    if (tid < 64) {
        const float* D = &sy[tid * 8];
        float h1 = D[0], h2 = D[1], v1 = D[2], v2 = D[3];
        float d1 = D[4], d2 = D[5], d3 = D[6], d4 = D[7];
        float ctr = -(h1 + h2 + v1 + v2);
        float* kp = ker + (b * 64 + tid) * 9;
        kp[0] = d1; kp[1] = v2;  kp[2] = d2;
        kp[3] = h2; kp[4] = ctr; kp[5] = h1;
        kp[6] = d4; kp[7] = v1;  kp[8] = d3;
    }
}

// ---------------------------------------------------------------------------
// convC: bc-conv2 on t1 (WMMA) + dynamic 3x3 conv on xs, ReLU both, sum.
// t1 is channel-innermost -> halo tile staged with GLOBAL_LOAD_ASYNC_TO_LDS
// (b128 per lane, ASYNCcnt tracked) when available.
// ---------------------------------------------------------------------------
__global__ __launch_bounds__(256)
void convC_kernel(const _Float16* __restrict__ t1,   // (b,128,128,64) f16
                  const float* __restrict__ xs,
                  const float* __restrict__ ker,
                  const float* __restrict__ bc_b2,
                  const _Float16* __restrict__ w16bc2,  // [64][576]
                  float* __restrict__ out)
{
    __shared__ _Float16 sm[3 * 130 * 64];
    __shared__ float sxs[3 * 130];

    const int b   = blockIdx.y;
    const int y   = blockIdx.x;
    const int tid = threadIdx.x;
    const int wid = tid >> 5;
    const int lane = tid & 31;

    // ---- stage t1 halo tile: 16-byte chunks, global -> LDS ----
    // chunk q within a row: col = q>>3 (0..129), part = q&7 (8 f16 each)
    #pragma unroll
    for (int row = 0; row < 3; ++row) {
        const int yy = y + row - 1;
        const bool rowok = (yy >= 0 && yy < HH);
        const _Float16* src = t1 + ((size_t)(b * HH + (rowok ? yy : 0)) * WW) * 64;
        for (int q = tid; q < 1040; q += 256) {
            const int col  = q >> 3;
            const int part = q & 7;
            const int xx   = col - 1;
            const int loff = (row * 130 + col) * 64 + part * 8;
            if (rowok && xx >= 0 && xx < WW) {
#if ASYNC_LDS
                __builtin_amdgcn_global_load_async_to_lds_b128(
                    (AS1 v4i*)(src + (size_t)xx * 64 + part * 8),
                    (AS3 v4i*)(&sm[loff]),
                    0, 0);
#else
                *(v8h*)(&sm[loff]) = *(const v8h*)(src + (size_t)xx * 64 + part * 8);
#endif
            } else {
                v8h z = {};
                *(v8h*)(&sm[loff]) = z;
            }
        }
    }
    // stage xs 3-row halo strip (tiny, synchronous)
    for (int i = tid; i < 3 * 130; i += 256) {
        int row = i / 130, col = i % 130;
        int yy = y + row - 1, xx = col - 1;
        float v = 0.0f;
        if (yy >= 0 && yy < HH && xx >= 0 && xx < WW)
            v = xs[((size_t)b * HH + yy) * WW + xx];
        sxs[i] = v;
    }
#if ASYNC_LDS
    wait_async0();
#endif
    __syncthreads();

    const int mbase   = (wid & 3) * 16;
    const int nhalf   = (wid >> 2) * 64;
    const int halfsel = lane >> 4;
    const int nloc    = lane & 15;

    v8f acc[4] = {};

    const int o  = mbase + nloc;
    const int K0 = halfsel ? 16 : 0;
    const _Float16* wp = w16bc2 + (size_t)o * KTOT;
    __builtin_prefetch(wp, 0, 1);

    #pragma unroll 2
    for (int s = 0; s < KSTEPS; ++s) {
        const int kb = s * 32;
        const int r  = s >> 1;
        const int dy = r / 3, dx = r % 3;
        const int c0 = ((s & 1) << 5) + (halfsel << 4);

        union { v16h v; v8h h[2]; } af;
        af.h[0] = *(const v8h*)(wp + kb + K0);
        af.h[1] = *(const v8h*)(wp + kb + K0 + 8);

        #pragma unroll
        for (int t = 0; t < 4; ++t) {
            int px = nhalf + t * 16 + nloc;
            const v16h bfrag = *(const v16h*)(&sm[(dy * 130 + px + dx) * 64 + c0]);
            acc[t] = __builtin_amdgcn_wmma_f32_16x16x32_f16(
                false, af.v, false, bfrag, (short)0, acc[t], false, false);
        }
    }

    const int chbase = mbase + halfsel * 8;
    #pragma unroll
    for (int j = 0; j < 8; ++j) {
        int ch = chbase + j;
        float bias = bc_b2[ch];
        const float* kp = ker + (b * COUT + ch) * 9;
        float k0 = kp[0], k1 = kp[1], k2 = kp[2];
        float k3 = kp[3], k4 = kp[4], k5 = kp[5];
        float k6 = kp[6], k7 = kp[7], k8 = kp[8];
        float* dst = out + (((size_t)(b * COUT + ch) * HH + y) * WW);
        #pragma unroll
        for (int t = 0; t < 4; ++t) {
            int px = nhalf + t * 16 + nloc;
            float base = fmaxf(acc[t][j] + bias, 0.0f);
            float dyn = k0 * sxs[px]       + k1 * sxs[px + 1]       + k2 * sxs[px + 2]
                      + k3 * sxs[130 + px] + k4 * sxs[130 + px + 1] + k5 * sxs[130 + px + 2]
                      + k6 * sxs[260 + px] + k7 * sxs[260 + px + 1] + k8 * sxs[260 + px + 2];
            dst[px] = base + fmaxf(dyn, 0.0f);
        }
    }
}

// ---------------------------------------------------------------------------
// launch
// ---------------------------------------------------------------------------
extern "C" void kernel_launch(void* const* d_in, const int* in_sizes, int n_in,
                              void* d_out, int out_size, void* d_ws, size_t ws_size,
                              hipStream_t stream)
{
    const float* x     = (const float*)d_in[0];
    const float* dn_w1 = (const float*)d_in[1];
    const float* dn_b1 = (const float*)d_in[2];
    const float* dn_w2 = (const float*)d_in[3];
    const float* dn_b2 = (const float*)d_in[4];
    const float* bc_w1 = (const float*)d_in[5];
    const float* bc_b1 = (const float*)d_in[6];
    const float* bc_w2 = (const float*)d_in[7];
    const float* bc_b2 = (const float*)d_in[8];

    char* ws = (char*)d_ws;
    // workspace layout (bytes)
    _Float16* t1  = (_Float16*)(ws);                      // 16*128*128*64*2 = 32 MiB (b,y,x,c)
    float*    xs  = (float*)(ws + 33554432);              // 16*128*128*4   = 1 MiB
    float*    gap = (float*)(ws + 34603008);              // 16*64*4
    float*    ker = (float*)(ws + 34607104);              // 16*64*9*4
    _Float16* w16 = (_Float16*)(ws + 34643968);           // 3 * 64*576*2

    prep_kernel<<<432, 256, 0, stream>>>(dn_w1, bc_w1, bc_w2, w16, gap);

    dim3 grid(HH, BATCH);
    convA_kernel<<<grid, 256, 0, stream>>>(x, dn_b1, bc_b1, w16, t1, xs, gap);
    head_kernel<<<BATCH, 256, 0, stream>>>(dn_w2, dn_b2, gap, ker);
    convC_kernel<<<grid, 256, 0, stream>>>(t1, xs, ker, bc_b2,
                                           w16 + 2 * COUT * KTOT, (float*)d_out);
}
